// MPLayer_60636348285179
// MI455X (gfx1250) — matrix operand order repaired and checked by reading, not compiled, not run on hardware
//
#include <hip/hip_runtime.h>
#include <math.h>

// ---------------------------------------------------------------------------
// CDNA5 (gfx1250) CGConv layer: bf16 WMMA GEMMs + f32 atomics scatter.
// ---------------------------------------------------------------------------

typedef __attribute__((ext_vector_type(16))) __bf16        v16bf;
typedef __attribute__((ext_vector_type(8)))  __bf16        v8bf;
typedef __attribute__((ext_vector_type(4)))  __bf16        v4bf;
typedef __attribute__((ext_vector_type(2)))  __bf16        v2bf;
typedef __attribute__((ext_vector_type(8)))  float         v8f;

#define ATOMF 64
#define ZDIM  192

// ---- float pair -> 2 x bf16 vector (stay in bf16-vector domain!) ----------
#if __has_builtin(__builtin_amdgcn_cvt_pk_bf16_f32)
__device__ __forceinline__ v2bf pk2v(float a, float b) {
    auto r = __builtin_amdgcn_cvt_pk_bf16_f32(a, b);   // elem0 = a, elem1 = b
    v2bf out; __builtin_memcpy(&out, &r, 4);
    return out;
}
#else
__device__ __forceinline__ v2bf pk2v(float a, float b) {
    v2bf o; o[0] = (__bf16)a; o[1] = (__bf16)b; return o;
}
#endif

// scalar f32 -> bf16 (RNE) for the tiny one-shot weight prep
__device__ __forceinline__ unsigned short f2bf_u(float f) {
    unsigned u = __builtin_bit_cast(unsigned, f);
    unsigned r = u + 0x7FFFu + ((u >> 16) & 1u);
    return (unsigned short)(r >> 16);
}
__device__ __forceinline__ float sigmoidf_(float x) {
    return 1.0f / (1.0f + __expf(-x));
}

// ---- build bf16 A fragment (16x32 tile, K-block t) from 3 z-regions -------
// A layout used here (contiguous K, B loads use the same order): lane L holds row L%16;
//   V0..V7 = K [kb, kb+16), kb = (L>=16)?16:0.
// 8-element chunks are 64-aligned-region safe (regions are 64 floats wide).
__device__ __forceinline__ v16bf build_A(const float* zp0, const float* zp1,
                                         const float* zp2, int t, int hi) {
    int c0 = t * 32 + (hi ? 16 : 0);
    int c1 = c0 + 8;
    const float* p0 = (c0 < 64 ? zp0 : (c0 < 128 ? zp1 : zp2)) + (c0 & 63);
    const float* p1 = (c1 < 64 ? zp0 : (c1 < 128 ? zp1 : zp2)) + (c1 & 63);
    float4 x0 = *(const float4*)(p0);
    float4 x1 = *(const float4*)(p0 + 4);
    float4 y0 = *(const float4*)(p1);
    float4 y1 = *(const float4*)(p1 + 4);
    v2bf a0 = pk2v(x0.x, x0.y), a1 = pk2v(x0.z, x0.w);
    v2bf a2 = pk2v(x1.x, x1.y), a3 = pk2v(x1.z, x1.w);
    v2bf b0 = pk2v(y0.x, y0.y), b1 = pk2v(y0.z, y0.w);
    v2bf b2 = pk2v(y1.x, y1.y), b3 = pk2v(y1.z, y1.w);
    v4bf q0 = __builtin_shufflevector(a0, a1, 0, 1, 2, 3);
    v4bf q1 = __builtin_shufflevector(a2, a3, 0, 1, 2, 3);
    v4bf q2 = __builtin_shufflevector(b0, b1, 0, 1, 2, 3);
    v4bf q3 = __builtin_shufflevector(b2, b3, 0, 1, 2, 3);
    v8bf h0 = __builtin_shufflevector(q0, q1, 0, 1, 2, 3, 4, 5, 6, 7);
    v8bf h1 = __builtin_shufflevector(q2, q3, 0, 1, 2, 3, 4, 5, 6, 7);
    return __builtin_shufflevector(h0, h1, 0, 1, 2, 3, 4, 5, 6, 7,
                                   8, 9, 10, 11, 12, 13, 14, 15);
}

// ---------------------------------------------------------------------------
// Kernel: weight prep (f32 -> bf16, WMMA-B friendly row-major, zero padding)
// ws layout (ushort elems): wf[64*192] | ws[64*192] | w1[16*192] | w2[64*32]
// ---------------------------------------------------------------------------
__global__ void prep_kernel(const float* __restrict__ lfw, const float* __restrict__ lsw,
                            const float* __restrict__ f1w, const float* __restrict__ f2w,
                            unsigned short* __restrict__ wf, unsigned short* __restrict__ wsm,
                            unsigned short* __restrict__ w1, unsigned short* __restrict__ w2) {
    int i = blockIdx.x * blockDim.x + threadIdx.x;
    if (i < 64 * 192) { wf[i] = f2bf_u(lfw[i]); wsm[i] = f2bf_u(lsw[i]); }
    if (i < 16 * 192) {                       // fc1: pad out-cols 14,15 with zeros
        int r = i / 192;
        w1[i] = (r < 14) ? f2bf_u(f1w[i]) : (unsigned short)0;
    }
    if (i < 64 * 32) {                        // fc2: pad K 14..31 with zeros
        int r = i >> 5, k = i & 31;
        w2[i] = (k < 14) ? f2bf_u(f2w[r * 14 + k]) : (unsigned short)0;
    }
}

__global__ void copy_kernel(const float* __restrict__ a, float* __restrict__ o, int n) {
    int i = blockIdx.x * blockDim.x + threadIdx.x;
    if (i < n) o[i] = a[i];
}

// ---------------------------------------------------------------------------
// Kernel 1: gated message + scatter-add.  One wave = 16 edges.
// z = [atom[dst] | atom[src] | edge_fea];  msg = sig(zWf+bf)*softplus(zWs+bs)*gauss
// ---------------------------------------------------------------------------
__global__ void __launch_bounds__(256) msg_kernel(
    const float* __restrict__ atom, const int* __restrict__ srci, const int* __restrict__ dsti,
    const float* __restrict__ efea, const float* __restrict__ dist,
    const unsigned short* __restrict__ wf, const unsigned short* __restrict__ wsm,
    const float* __restrict__ biasf, const float* __restrict__ biass,
    float* __restrict__ atom_out, int ntiles) {
    int wave = blockIdx.x * (blockDim.x >> 5) + (threadIdx.x >> 5);
    if (wave >= ntiles) return;
    int lane = threadIdx.x & 31, l16 = lane & 15, hi = lane >> 4;
    int e0 = wave * 16;

    // A-side gather pointers (this lane's z row)
    int ea = e0 + l16;
    const float* zp0 = atom + (long)dsti[ea] * ATOMF;   // x_i
    const float* zp1 = atom + (long)srci[ea] * ATOMF;   // x_j
    const float* zp2 = efea + (long)ea * ATOMF;

    // C-side per-row data (row = v + 8*hi)
    int dstC[8]; float gf[8];
#pragma unroll
    for (int v = 0; v < 8; ++v) {
        int e = e0 + v + hi * 8;
        dstC[v] = dsti[e];
        float d = dist[e];
        gf[v] = __expf(-d * d * (1.0f / 18.0f));
    }

    v8f accF[4], accS[4];
#pragma unroll
    for (int j = 0; j < 4; ++j) { accF[j] = (v8f)0.0f; accS[j] = (v8f)0.0f; }

#pragma unroll
    for (int t = 0; t < 6; ++t) {
        v16bf A = build_A(zp0, zp1, zp2, t, hi);
        int kb2 = t * 32 + (hi ? 16 : 0);
#pragma unroll
        for (int j = 0; j < 4; ++j) {
            int col = j * 16 + l16;
            v16bf Bf = *(const v16bf*)(wf  + (long)col * ZDIM + kb2);
            v16bf Bs = *(const v16bf*)(wsm + (long)col * ZDIM + kb2);
            accF[j] = __builtin_amdgcn_wmma_f32_16x16x32_bf16(false, A, false, Bf,
                                                              (short)0, accF[j], false, false);
            accS[j] = __builtin_amdgcn_wmma_f32_16x16x32_bf16(false, A, false, Bs,
                                                              (short)0, accS[j], false, false);
        }
    }

#pragma unroll
    for (int j = 0; j < 4; ++j) {
        int col = j * 16 + l16;
        float bf_ = biasf[col], bs_ = biass[col];
#pragma unroll
        for (int v = 0; v < 8; ++v) {
            float f = accF[j][v] + bf_;
            float s = accS[j][v] + bs_;
            float sp = (s > 20.0f) ? s : log1pf(__expf(s));  // softplus
            float m = sigmoidf_(f) * sp * gf[v];
            __hip_atomic_fetch_add(atom_out + (long)dstC[v] * ATOMF + col, m,
                                   __ATOMIC_RELAXED, __HIP_MEMORY_SCOPE_AGENT);
        }
    }
}

// ---------------------------------------------------------------------------
// Kernel 2: edge-update MLP. z2 = [out[src] | out[dst] | edge_fea]
// h = silu(z2 W1 + b1) [14, padded 16];  edge_out = silu(h W2 + b2) [64]
// ---------------------------------------------------------------------------
__global__ void __launch_bounds__(256) edge_kernel(
    const float* __restrict__ atom_out, const int* __restrict__ srci, const int* __restrict__ dsti,
    const float* __restrict__ efea,
    const unsigned short* __restrict__ w1, const unsigned short* __restrict__ w2,
    const float* __restrict__ b1v, const float* __restrict__ b2v,
    float* __restrict__ edge_out, int ntiles) {
    __shared__ __align__(64) unsigned short hbuf[8 * 16 * 32];  // per-wave 16x32 bf16 h tile
    int wavein = threadIdx.x >> 5;
    int wave = blockIdx.x * (blockDim.x >> 5) + wavein;
    if (wave >= ntiles) return;
    int lane = threadIdx.x & 31, l16 = lane & 15, hi = lane >> 4;
    int e0 = wave * 16;

    int ea = e0 + l16;
    const float* zp0 = atom_out + (long)srci[ea] * ATOMF;   // x_row
    const float* zp1 = atom_out + (long)dsti[ea] * ATOMF;   // x_col
    const float* zp2 = efea + (long)ea * ATOMF;

    v8f h = (v8f)0.0f;
#pragma unroll
    for (int t = 0; t < 6; ++t) {
        v16bf A = build_A(zp0, zp1, zp2, t, hi);
        int kb2 = t * 32 + (hi ? 16 : 0);
        v16bf B1 = *(const v16bf*)(w1 + (long)l16 * ZDIM + kb2);
        h = __builtin_amdgcn_wmma_f32_16x16x32_bf16(false, A, false, B1,
                                                    (short)0, h, false, false);
    }

    // bias + SiLU, stage h tile (16x32, K padded with zeros) in LDS as bf16
    float b1 = (l16 < 14) ? b1v[l16] : 0.0f;
    unsigned short* my = hbuf + wavein * 16 * 32;
#pragma unroll
    for (int v = 0; v < 8; ++v) {
        int row = v + hi * 8;
        float x = h[v] + b1;
        float y = x * sigmoidf_(x);
        my[row * 32 + l16]      = f2bf_u(y);   // K = l16  (cols 14,15 are silu(0)=0)
        my[row * 32 + 16 + l16] = 0;           // K pad 16..31
    }
    __builtin_amdgcn_wave_barrier();
    asm volatile("s_wait_dscnt 0" ::: "memory");   // DS RAW fence (wave-local tile)
    __builtin_amdgcn_wave_barrier();

    // re-layout C-fragment -> A-fragment via LDS
    int kb = hi ? 16 : 0;
    v8bf alo = *(const v8bf*)(my + l16 * 32 + kb);
    v8bf ahi = *(const v8bf*)(my + l16 * 32 + kb + 8);
    v16bf A2 = __builtin_shufflevector(alo, ahi, 0, 1, 2, 3, 4, 5, 6, 7,
                                       8, 9, 10, 11, 12, 13, 14, 15);

#pragma unroll
    for (int j = 0; j < 4; ++j) {
        int col = j * 16 + l16;
        v16bf B2 = *(const v16bf*)(w2 + (long)col * 32 + (hi ? 16 : 0));
        v8f acc = (v8f)0.0f;
        acc = __builtin_amdgcn_wmma_f32_16x16x32_bf16(false, A2, false, B2,
                                                      (short)0, acc, false, false);
        float b2 = b2v[col];
#pragma unroll
        for (int v = 0; v < 8; ++v) {
            long e = e0 + v + hi * 8;
            float x = acc[v] + b2;
            edge_out[e * ATOMF + col] = x * sigmoidf_(x);
        }
    }
}

// ---------------------------------------------------------------------------
extern "C" void kernel_launch(void* const* d_in, const int* in_sizes, int n_in,
                              void* d_out, int out_size, void* d_ws, size_t ws_size,
                              hipStream_t stream) {
    const float* atom = (const float*)d_in[0];
    const int*   eidx = (const int*)d_in[1];
    const float* efea = (const float*)d_in[2];
    const float* dist = (const float*)d_in[4];
    const float* lfw  = (const float*)d_in[6];
    const float* lfb  = (const float*)d_in[7];
    const float* lsw  = (const float*)d_in[8];
    const float* lsb  = (const float*)d_in[9];
    const float* f1w  = (const float*)d_in[10];
    const float* f1b  = (const float*)d_in[11];
    const float* f2w  = (const float*)d_in[12];
    const float* f2b  = (const float*)d_in[13];

    int  nnodes = in_sizes[0] / ATOMF;
    long E      = in_sizes[1] / 2;
    const int* src = eidx;
    const int* dst = eidx + E;

    float* atom_out = (float*)d_out;
    float* edge_out = atom_out + (long)nnodes * ATOMF;

    unsigned short* wf  = (unsigned short*)d_ws;
    unsigned short* wsm = wf  + 64 * 192;
    unsigned short* w1  = wsm + 64 * 192;
    unsigned short* w2  = w1  + 16 * 192;

    // 1) weight conversion / padding (covers 64*192 = 12288 indices)
    prep_kernel<<<(64 * 192 + 255) / 256, 256, 0, stream>>>(lfw, lsw, f1w, f2w,
                                                            wf, wsm, w1, w2);
    // 2) residual init: atom_out = atom_fea
    int ncopy = nnodes * ATOMF;
    copy_kernel<<<(ncopy + 255) / 256, 256, 0, stream>>>(atom, atom_out, ncopy);

    // 3) message + scatter (one wave / 16 edges, 8 waves / block)
    int ntiles = (int)((E + 15) / 16);
    int blocks = (ntiles + 7) / 8;
    msg_kernel<<<blocks, 256, 0, stream>>>(atom, src, dst, efea, dist,
                                           wf, wsm, lfb, lsb, atom_out, ntiles);
    // 4) edge MLP
    edge_kernel<<<blocks, 256, 0, stream>>>(atom_out, src, dst, efea,
                                            w1, w2, f1b, f2b, edge_out, ntiles);
}
